// SwinMambaBlock_8641474200043
// MI455X (gfx1250) — compile-verified, matrix operand order checked by reading, not run on hardware
//
#include <hip/hip_runtime.h>
#include <hip/hip_bf16.h>

typedef __attribute__((ext_vector_type(16))) __bf16 v16bf;
typedef __attribute__((ext_vector_type(8)))  float  v8f;

#define DIM   384
#define HID   1536
#define NTOK  50176   // 1024 windows * 49 tokens == 16*56*56 pixels

// ---------------------------------------------------------------------------
// WMMA helpers (ISA 7.12.2 layouts, wave32)
// ---------------------------------------------------------------------------
__device__ __forceinline__ v8f wmma_bf16(v16bf a, v16bf b, v8f c) {
  // emits v_wmma_f32_16x16x32_bf16
  return __builtin_amdgcn_wmma_f32_16x16x32_bf16(
      /*neg_a=*/false, a, /*neg_b=*/false, b,
      /*c_mod=*/(short)0, c, /*reuse_a=*/false, /*reuse_b=*/false);
}

// Unified fragment loader. Works for:
//  - A operand from row-major [m][k] tile (s = tile base, ld = k-stride)
//  - B operand from TRANSPOSED [n][k] tile (s = tile base + nbase*ld)
// Lane L: row/col = L&15, k-half = 8*(L>>4); elems 0..7 -> k+0..7,
// elems 8..15 -> k+16..23.  Two 16-byte loads per fragment.
union FragCast { v16bf v; uint4 q[2]; };
__device__ __forceinline__ v16bf frag_ld(const __bf16* s, int ld) {
  const int lane = threadIdx.x & 31;
  const __bf16* p = s + (lane & 15) * ld + ((lane >> 4) << 3);
  FragCast f;
  f.q[0] = *(const uint4*)p;          // k + 0..7
  f.q[1] = *(const uint4*)(p + 16);   // k + 16..23
  return f.v;
}
// C/D: element r of lane L -> row = r + 8*(L>>4), col = L&15

// ---------------------------------------------------------------------------
// 0) fp32 -> bf16 transposed conversion: src (R x C) -> dst (C x R)
//    Produces [n][k] weight layouts so B fragments are k-contiguous.
// ---------------------------------------------------------------------------
__global__ void cvt_bf16_t_kernel(const float* __restrict__ src,
                                  __bf16* __restrict__ dst, int R, int C) {
  int i = blockIdx.x * blockDim.x + threadIdx.x;
  if (i < R * C) {
    int r = i / C, c = i - r * C;
    dst[(size_t)c * R + r] = (__bf16)src[i];
  }
}

// ---------------------------------------------------------------------------
// 1) LayerNorm (+ optional 7x7 window-partition gather) -> bf16 rows
// ---------------------------------------------------------------------------
__global__ void ln_kernel(const float* __restrict__ x,
                          const float* __restrict__ gw,
                          const float* __restrict__ gb,
                          __bf16* __restrict__ out, int windowed) {
  const int tk   = blockIdx.x * 8 + (threadIdx.x >> 5);
  const int lane = threadIdx.x & 31;
  int srow;
  if (windowed) {
    const int win = tk / 49, t = tk - win * 49;
    const int b = win >> 6, wh = (win >> 3) & 7, ww = win & 7;
    const int i = t / 7, j = t - i * 7;
    srow = (b * 56 + wh * 7 + i) * 56 + (ww * 7 + j);
  } else {
    srow = tk;
  }
  const float* p = x + (size_t)srow * DIM;
  float v[12], s = 0.f, ss = 0.f;
#pragma unroll
  for (int u = 0; u < 12; ++u) {
    v[u] = p[lane + 32 * u];
    s += v[u]; ss += v[u] * v[u];
  }
#pragma unroll
  for (int o = 16; o > 0; o >>= 1) {
    s  += __shfl_xor(s, o, 32);
    ss += __shfl_xor(ss, o, 32);
  }
  const float mean = s * (1.f / DIM);
  const float rstd = rsqrtf(ss * (1.f / DIM) - mean * mean + 1e-5f);
  __bf16* q = out + (size_t)tk * DIM;
#pragma unroll
  for (int u = 0; u < 12; ++u) {
    const int c = lane + 32 * u;
    q[c] = (__bf16)((v[u] - mean) * rstd * gw[c] + gb[c]);
  }
}

// ---------------------------------------------------------------------------
// 2/4) GEMM: (NTOK x 384) x (384 x 384), A row-major bf16, B given TRANSPOSED
//    [n][k].  BM=128 BN=64 BK=32, 8 waves x (32x32).
//    EPI=0: plain row-major f32 store (xB)
//    EPI=1: window-reverse scatter to (B,56,56,C) f32 (xr)
// ---------------------------------------------------------------------------
template <int EPI>
__global__ void gemm384_kernel(const __bf16* __restrict__ A,
                               const __bf16* __restrict__ Bt,  // [384 n][384 k]
                               float* __restrict__ out) {
  __shared__ __attribute__((aligned(16))) __bf16 As[128 * 32];  // [m][k]
  __shared__ __attribute__((aligned(16))) __bf16 Bs[64 * 32];   // [n][k]
  const int tid = threadIdx.x;
  const int lane = tid & 31;
  const int w  = tid >> 5;
  const int mw = w >> 1, nw = w & 1;
  const int m0 = blockIdx.y * 128;
  const int n0 = blockIdx.x * 64;

  v8f c00 = {}, c01 = {}, c10 = {}, c11 = {};
  for (int kb = 0; kb < 12; ++kb) {
    const int k0 = kb * 32;
    __syncthreads();
    {  // stage A tile 128x32 (row-major)
      const int row = tid >> 1, half = (tid & 1) * 16;
      const uint4* src = (const uint4*)(A + (size_t)(m0 + row) * DIM + k0 + half);
      uint4* dst = (uint4*)(As + row * 32 + half);
      dst[0] = src[0]; dst[1] = src[1];
    }
    {  // stage B tile as [n][k] straight from transposed weight
      const int row = tid >> 2, c8 = (tid & 3) * 8;  // 64 rows x 4 chunks
      *(uint4*)(Bs + row * 32 + c8) =
          *(const uint4*)(Bt + (size_t)(n0 + row) * DIM + k0 + c8);
    }
    __syncthreads();
    const v16bf a0 = frag_ld(As + (mw * 32) * 32, 32);
    const v16bf a1 = frag_ld(As + (mw * 32 + 16) * 32, 32);
    const v16bf b0 = frag_ld(Bs + (nw * 32) * 32, 32);
    const v16bf b1 = frag_ld(Bs + (nw * 32 + 16) * 32, 32);
    c00 = wmma_bf16(a0, b0, c00);
    c01 = wmma_bf16(a0, b1, c01);
    c10 = wmma_bf16(a1, b0, c10);
    c11 = wmma_bf16(a1, b1, c11);
  }
  const int crow0 = 8 * (lane >> 4);
  const int ccol  = lane & 15;
  const v8f cf[4] = {c00, c01, c10, c11};
#pragma unroll
  for (int ti = 0; ti < 2; ++ti)
#pragma unroll
    for (int tj = 0; tj < 2; ++tj) {
      const v8f c = cf[ti * 2 + tj];
      const int mrow = m0 + mw * 32 + ti * 16;
      const int ncol = n0 + nw * 32 + tj * 16 + ccol;
#pragma unroll
      for (int r = 0; r < 8; ++r) {
        const int row = mrow + crow0 + r;
        if (EPI == 0) {
          out[(size_t)row * DIM + ncol] = c[r];
        } else {
          const int win = row / 49, t = row - win * 49;
          const int b = win >> 6, wh = (win >> 3) & 7, ww = win & 7;
          const int i = t / 7, j = t - i * 7;
          const int p = (b * 56 + wh * 7 + i) * 56 + (ww * 7 + j);
          out[(size_t)p * DIM + ncol] = c[r];
        }
      }
    }
}

// ---------------------------------------------------------------------------
// 3) Mamba scan: h_t = xB_t + h_{t-1} @ A, out h_t as bf16 tokens.
//    16 windows per block; recurrent state kept as bf16 in LDS.
//    A is provided TRANSPOSED [n][k]; k-slice (384x32) staged per k-step.
// ---------------------------------------------------------------------------
__global__ void mamba_scan_kernel(const float* __restrict__ xB,
                                  const __bf16* __restrict__ At,  // [n][k]
                                  __bf16* __restrict__ hout) {
  __shared__ __attribute__((aligned(16))) __bf16 hb[16 * DIM];   // state [m][k]
  __shared__ __attribute__((aligned(16))) __bf16 As[DIM * 32];   // [n][k-slice]
  const int tid  = threadIdx.x;
  const int lane = tid & 31;
  const int w    = tid >> 5;
  const int win0 = blockIdx.x * 16;
  const int nbase = w * 48;

  for (int i = tid; i < 16 * DIM; i += 256) hb[i] = (__bf16)0.f;

  for (int t = 0; t < 49; ++t) {
    v8f c0 = {}, c1 = {}, c2 = {};
    for (int kb = 0; kb < 12; ++kb) {
      __syncthreads();  // prior reads of As / writes of hb complete
      {  // stage At[:, kb*32 .. +32) -> As[384][32]; 1536 16B-chunks / 256 thr
#pragma unroll
        for (int u = 0; u < 6; ++u) {
          const int id = tid + 256 * u;
          const int row = id >> 2, c8 = (id & 3) * 8;
          *(uint4*)(As + row * 32 + c8) =
              *(const uint4*)(At + (size_t)row * DIM + kb * 32 + c8);
        }
      }
      __syncthreads();
      const v16bf a = frag_ld(hb + kb * 32, DIM);
      c0 = wmma_bf16(a, frag_ld(As + (nbase)      * 32, 32), c0);
      c1 = wmma_bf16(a, frag_ld(As + (nbase + 16) * 32, 32), c1);
      c2 = wmma_bf16(a, frag_ld(As + (nbase + 32) * 32, 32), c2);
    }
    __syncthreads();  // all waves done reading hb for this step
    const int crow0 = 8 * (lane >> 4);
    const int ccol  = lane & 15;
    const v8f cc[3] = {c0, c1, c2};
#pragma unroll
    for (int jt = 0; jt < 3; ++jt) {
      const int col = nbase + jt * 16 + ccol;
#pragma unroll
      for (int r = 0; r < 8; ++r) {
        const int row = crow0 + r;
        const size_t tok = (size_t)(win0 + row) * 49 + t;
        const float v = cc[jt][r] + xB[tok * DIM + col];
        const __bf16 vb = (__bf16)v;
        hb[row * DIM + col] = vb;      // next-step state
        hout[tok * DIM + col] = vb;    // token output for h @ Cm
      }
    }
  }
}

// ---------------------------------------------------------------------------
// 5) Fused MLP: 16 tokens/block. GEMM1 (384->1536) + bias + exact GELU into
//    LDS (bf16), then GEMM2 (1536->384) + bias + residual -> f32 out.
//    W1/W2 provided transposed [n][k]; L2-resident, fragments read directly.
// ---------------------------------------------------------------------------
__global__ void mlp_kernel(const __bf16* __restrict__ X,
                           const __bf16* __restrict__ W1t,  // [1536][384]
                           const float* __restrict__ b1,
                           const __bf16* __restrict__ W2t,  // [384][1536]
                           const float* __restrict__ b2,
                           const float* __restrict__ xr,
                           float* __restrict__ out) {
  __shared__ __attribute__((aligned(16))) __bf16 Xs[16 * DIM];
  __shared__ __attribute__((aligned(16))) __bf16 Act[16 * HID];
  const int tid  = threadIdx.x;
  const int lane = tid & 31;
  const int w    = tid >> 5;
  const int p0   = blockIdx.x * 16;

  {  // stage 16x384 token tile: 24 bf16 per thread
    const uint4* src = (const uint4*)(X + (size_t)p0 * DIM + tid * 24);
    uint4* dst = (uint4*)(Xs + tid * 24);
    dst[0] = src[0]; dst[1] = src[1]; dst[2] = src[2];
  }
  __syncthreads();
  const int crow0 = 8 * (lane >> 4);
  const int ccol  = lane & 15;

  // GEMM1 + GELU -> Act (LDS)
  for (int np = 0; np < 12; ++np) {
    const int colb = np * 128 + w * 16;
    v8f c = {};
    for (int kb = 0; kb < 12; ++kb) {
      const v16bf a = frag_ld(Xs + kb * 32, DIM);
      const v16bf b = frag_ld(W1t + (size_t)colb * DIM + kb * 32, DIM);
      c = wmma_bf16(a, b, c);
    }
#pragma unroll
    for (int r = 0; r < 8; ++r) {
      const int row = crow0 + r;
      const int col = colb + ccol;
      float v = c[r] + b1[col];
      v = 0.5f * v * (1.0f + erff(v * 0.70710678118f));  // exact GELU
      Act[row * HID + col] = (__bf16)v;
    }
  }
  __syncthreads();

  // GEMM2 + bias + residual -> out
  for (int np = 0; np < 3; ++np) {
    const int colb = np * 128 + w * 16;
    v8f c = {};
    for (int kb = 0; kb < 48; ++kb) {
      const v16bf a = frag_ld(Act + kb * 32, HID);
      const v16bf b = frag_ld(W2t + (size_t)colb * HID + kb * 32, HID);
      c = wmma_bf16(a, b, c);
    }
#pragma unroll
    for (int r = 0; r < 8; ++r) {
      const int row = crow0 + r;
      const int col = colb + ccol;
      const size_t idx = (size_t)(p0 + row) * DIM + col;
      out[idx] = c[r] + b2[col] + xr[idx];
    }
  }
}

// ---------------------------------------------------------------------------
// Launch
// ---------------------------------------------------------------------------
extern "C" void kernel_launch(void* const* d_in, const int* in_sizes, int n_in,
                              void* d_out, int out_size, void* d_ws, size_t ws_size,
                              hipStream_t stream) {
  const float* x     = (const float*)d_in[0];
  const float* Af    = (const float*)d_in[1];
  const float* Bmf   = (const float*)d_in[2];
  const float* Cmf   = (const float*)d_in[3];
  const float* ln1w  = (const float*)d_in[4];
  const float* ln1b  = (const float*)d_in[5];
  const float* ln2w  = (const float*)d_in[6];
  const float* ln2b  = (const float*)d_in[7];
  const float* W1f   = (const float*)d_in[8];
  const float* b1    = (const float*)d_in[9];
  const float* W2f   = (const float*)d_in[10];
  const float* b2    = (const float*)d_in[11];
  float* out = (float*)d_out;

  char* ws = (char*)d_ws;
  // transposed bf16 weights ([n][k] layouts)
  __bf16* At   = (__bf16*)(ws + 0);          //  384x384
  __bf16* Bt   = (__bf16*)(ws + 294912);     //  384x384
  __bf16* Ct   = (__bf16*)(ws + 589824);     //  384x384
  __bf16* W1t  = (__bf16*)(ws + 884736);     // 1536x384
  __bf16* W2t  = (__bf16*)(ws + 2064384);    //  384x1536
  // activations (aliased: wln1/hbf share, xB/ln2bf share)
  __bf16* wln1  = (__bf16*)(ws + 3244032);   // 50176*384 bf16
  __bf16* hbf   = wln1;                      // reuse after xB GEMM consumed wln1
  float*  xB    = (float*)(ws + 41779200);   // 50176*384 f32
  __bf16* ln2bf = (__bf16*)(ws + 41779200);  // reuse after scan consumed xB
  float*  xr    = (float*)(ws + 118849536);  // 50176*384 f32

  // 0) weight conversion + transpose
  cvt_bf16_t_kernel<<<(DIM * DIM + 255) / 256, 256, 0, stream>>>(Af,  At,  DIM, DIM);
  cvt_bf16_t_kernel<<<(DIM * DIM + 255) / 256, 256, 0, stream>>>(Bmf, Bt,  DIM, DIM);
  cvt_bf16_t_kernel<<<(DIM * DIM + 255) / 256, 256, 0, stream>>>(Cmf, Ct,  DIM, DIM);
  cvt_bf16_t_kernel<<<(DIM * HID + 255) / 256, 256, 0, stream>>>(W1f, W1t, DIM, HID);
  cvt_bf16_t_kernel<<<(DIM * HID + 255) / 256, 256, 0, stream>>>(W2f, W2t, HID, DIM);

  // 1) LN1 + window partition
  ln_kernel<<<NTOK / 8, 256, 0, stream>>>(x, ln1w, ln1b, wln1, 1);

  // 2) xB = wln1 @ Bm
  gemm384_kernel<0><<<dim3(DIM / 64, NTOK / 128), 256, 0, stream>>>(wln1, Bt, xB);

  // 3) recurrent scan -> hbf (bf16 tokens)
  mamba_scan_kernel<<<64, 256, 0, stream>>>(xB, At, hbf);

  // 4) xr = window_reverse(h @ Cm)
  gemm384_kernel<1><<<dim3(DIM / 64, NTOK / 128), 256, 0, stream>>>(hbf, Ct, xr);

  // 5) LN2 (pixel-row order)
  ln_kernel<<<NTOK / 8, 256, 0, stream>>>(xr, ln2w, ln2b, ln2bf, 0);

  // 6) fused MLP + residual
  mlp_kernel<<<NTOK / 16, 256, 0, stream>>>(ln2bf, W1t, b1, W2t, b2, xr, out);
}
